// Attention_Action_1563368096230
// MI455X (gfx1250) — matrix operand order repaired and checked by reading, not compiled, not run on hardware
//
#include <hip/hip_runtime.h>
#include <hip/hip_bf16.h>

// ---------------------------------------------------------------------------
// MHA forward for MI455X (gfx1250, wave32, WMMA).
//   K1: qkv = x @ w_qkv   -> q (scaled by 0.125), k as [B,H,N,D] f16, v^T as [B,H,D,N] f16
//   K2: flash attention per (b,h), 64 q-rows per block, 32-key tiles, online softmax
//   K3: out = ctx @ w_out + b_out (f32 result)
// All matmuls lower to v_wmma_f32_16x16x32_f16 (f32 accumulate).
// ---------------------------------------------------------------------------

typedef __attribute__((ext_vector_type(16))) _Float16 v16h;
typedef __attribute__((ext_vector_type(8)))  _Float16 v8h;
typedef __attribute__((ext_vector_type(4)))  _Float16 v4h;
typedef __attribute__((ext_vector_type(8)))  float    v8f;

#define DIM        1024
#define NHEAD      16
#define HDIM       64
#define NB         4
#define NSEQ       1024
#define QKV_E      3072
#define LDSS       40            // LDS row stride in halves (80B, 16B-aligned rows)

static __device__ __forceinline__ v16h cat8(v8h a, v8h b) {
  return __builtin_shufflevector(a, b, 0,1,2,3,4,5,6,7,8,9,10,11,12,13,14,15);
}

static __device__ __forceinline__ v8f wmma16(v16h a, v16h b, v8f c) {
  return __builtin_amdgcn_wmma_f32_16x16x32_f16(false, a, false, b, (short)0, c,
                                                false, false);
}

// A fragment (16x32 f16) from LDS: row = lane&15, chunks at k=hi*8 and hi*8+16
static __device__ __forceinline__ v16h lds_afrag(const _Float16* base, int row, int hi) {
  const _Float16* rp = base + row * LDSS + hi * 8;
  return cat8(*(const v8h*)rp, *(const v8h*)(rp + 16));
}
// B fragment (32x16 f16) from LDS transposed tile [n][k]: col = lane&15, chunks at k=hi*8 and hi*8+16
static __device__ __forceinline__ v16h lds_bfrag(const _Float16* base, int col, int hi) {
  const _Float16* rp = base + col * LDSS + hi * 8;
  return cat8(*(const v8h*)rp, *(const v8h*)(rp + 16));
}

// ===========================================================================
// Kernel 1: qkv projection.  grid=(3072/128, 4096/128), block=256 (8 waves)
// ===========================================================================
__global__ __launch_bounds__(256) void k_qkv_gemm(
    const float* __restrict__ x, const float* __restrict__ w,
    _Float16* __restrict__ qb, _Float16* __restrict__ kb,
    _Float16* __restrict__ vtb) {
  __shared__ _Float16 As[128 * LDSS];   // x tile   [m][k]
  __shared__ _Float16 Bs[128 * LDSS];   // w tile   [e][k] (transposed)

  const int tid  = threadIdx.x;
  const int wave = tid >> 5, lane = tid & 31;
  const int l15  = lane & 15, hi = lane >> 4;
  const int waveM = wave >> 2, waveN = wave & 3;   // 2 x 4 wave grid
  const int Mtile = blockIdx.y * 128;
  const int Etile = blockIdx.x * 128;

  v8f acc[4][2];
  #pragma unroll
  for (int i = 0; i < 4; ++i)
    #pragma unroll
    for (int j = 0; j < 2; ++j) acc[i][j] = {};

  for (int k0 = 0; k0 < DIM; k0 += 32) {
    // stage X tile (f32 -> f16)
    #pragma unroll
    for (int p = 0; p < 4; ++p) {
      const int row = p * 32 + (tid >> 3);
      const int cg  = (tid & 7) * 4;
      float4 xv = *(const float4*)(x + (size_t)(Mtile + row) * DIM + k0 + cg);
      v4h hv = { (_Float16)xv.x, (_Float16)xv.y, (_Float16)xv.z, (_Float16)xv.w };
      *(v4h*)(As + row * LDSS + cg) = hv;
    }
    // stage W tile transposed (f32 -> f16)
    {
      const int kk = tid >> 5;
      const int e0 = (tid & 31) * 4;
      #pragma unroll
      for (int p = 0; p < 4; ++p) {
        const int kr = p * 8 + kk;
        float4 wv = *(const float4*)(w + (size_t)(k0 + kr) * QKV_E + Etile + e0);
        Bs[(e0 + 0) * LDSS + kr] = (_Float16)wv.x;
        Bs[(e0 + 1) * LDSS + kr] = (_Float16)wv.y;
        Bs[(e0 + 2) * LDSS + kr] = (_Float16)wv.z;
        Bs[(e0 + 3) * LDSS + kr] = (_Float16)wv.w;
      }
    }
    __syncthreads();

    v16h af[4], bf[2];
    #pragma unroll
    for (int ms = 0; ms < 4; ++ms)
      af[ms] = lds_afrag(As, waveM * 64 + ms * 16 + l15, hi);
    #pragma unroll
    for (int ns = 0; ns < 2; ++ns)
      bf[ns] = lds_bfrag(Bs, waveN * 32 + ns * 16 + l15, hi);
    #pragma unroll
    for (int ms = 0; ms < 4; ++ms)
      #pragma unroll
      for (int ns = 0; ns < 2; ++ns)
        acc[ms][ns] = wmma16(af[ms], bf[ns], acc[ms][ns]);
    __syncthreads();
  }

  // scatter epilogue: q (scaled), k as [B,H,N,D]; v transposed as [B,H,D,N]
  #pragma unroll
  for (int ns = 0; ns < 2; ++ns) {
    const int e = Etile + waveN * 32 + ns * 16 + l15;
    const int three = e >> 10, h = (e >> 6) & 15, d = e & 63;
    #pragma unroll
    for (int ms = 0; ms < 4; ++ms) {
      const int mb = Mtile + waveM * 64 + ms * 16 + hi * 8;
      const int b = mb >> 10, n0 = mb & 1023;
      const int bh = b * NHEAD + h;
      if (three == 0) {
        _Float16* qp = qb + ((size_t)bh * NSEQ + n0) * HDIM + d;
        #pragma unroll
        for (int r = 0; r < 8; ++r)
          qp[(size_t)r * HDIM] = (_Float16)(acc[ms][ns][r] * 0.125f);  // fold 1/sqrt(D)
      } else if (three == 1) {
        _Float16* kp = kb + ((size_t)bh * NSEQ + n0) * HDIM + d;
        #pragma unroll
        for (int r = 0; r < 8; ++r)
          kp[(size_t)r * HDIM] = (_Float16)acc[ms][ns][r];
      } else {
        v8h pv;
        #pragma unroll
        for (int r = 0; r < 8; ++r) pv[r] = (_Float16)acc[ms][ns][r];
        *(v8h*)(vtb + ((size_t)bh * HDIM + d) * NSEQ + n0) = pv;       // contiguous n
      }
    }
  }
}

// ===========================================================================
// Kernel 2: flash attention.  grid=(N/64, B*H), block=128 (4 waves x 16 rows)
// ===========================================================================
__global__ __launch_bounds__(128) void k_attn(
    const _Float16* __restrict__ qb, const _Float16* __restrict__ kb,
    const _Float16* __restrict__ vtb, const unsigned char* __restrict__ mask,
    _Float16* __restrict__ ctx) {
  __shared__ _Float16 Pt[4 * 16 * LDSS];   // per-wave P scratch (C-layout -> A-layout)

  const int tid = threadIdx.x;
  const int wave = tid >> 5, lane = tid & 31;
  const int l15 = lane & 15, hi = lane >> 4;
  const int bh = blockIdx.y, b = bh >> 4, h = bh & 15;
  const _Float16* qp = qb  + (size_t)bh * NSEQ * HDIM;
  const _Float16* kp = kb  + (size_t)bh * NSEQ * HDIM;
  const _Float16* vp = vtb + (size_t)bh * HDIM * NSEQ;
  const unsigned char* mp = mask + b * NSEQ;
  const int q0 = blockIdx.x * 64 + wave * 16;
  _Float16* myP = Pt + wave * 16 * LDSS;

  // Q A-fragments for both D halves (hoisted)
  v16h aq[2];
  #pragma unroll
  for (int ds = 0; ds < 2; ++ds) {
    const _Float16* rp = qp + (size_t)(q0 + l15) * HDIM + ds * 32 + hi * 8;
    aq[ds] = cat8(*(const v8h*)rp, *(const v8h*)(rp + 16));
  }

  float mrow[8], lrow[8];
  v8f acc[4];
  #pragma unroll
  for (int r = 0; r < 8; ++r) { mrow[r] = -1e30f; lrow[r] = 0.f; }
  #pragma unroll
  for (int d4 = 0; d4 < 4; ++d4) acc[d4] = {};

  for (int kt = 0; kt < NSEQ; kt += 32) {
    // K^T B-fragments: 2 key halves x 2 D-steps (two 16B chunks per lane, at k=hi*8 and hi*8+16)
    v16h bk0[2], bk1[2];
    #pragma unroll
    for (int ds = 0; ds < 2; ++ds) {
      const _Float16* r0 = kp + (size_t)(kt + l15) * HDIM + ds * 32 + hi * 8;
      bk0[ds] = cat8(*(const v8h*)r0, *(const v8h*)(r0 + 16));
      const _Float16* r1 = kp + (size_t)(kt + 16 + l15) * HDIM + ds * 32 + hi * 8;
      bk1[ds] = cat8(*(const v8h*)r1, *(const v8h*)(r1 + 16));
    }
    v8f s0 = {}, s1 = {};
    s0 = wmma16(aq[0], bk0[0], s0);  s0 = wmma16(aq[1], bk0[1], s0);
    s1 = wmma16(aq[0], bk1[0], s1);  s1 = wmma16(aq[1], bk1[1], s1);

    const float bias0 = mp[kt + l15]      ? 0.f : -1e30f;
    const float bias1 = mp[kt + 16 + l15] ? 0.f : -1e30f;

    // online softmax over this 32-key tile (rows r+8*hi, cols = lane&15)
    #pragma unroll
    for (int r = 0; r < 8; ++r) {
      float e0 = s0[r] + bias0;
      float e1 = s1[r] + bias1;
      float tm = fmaxf(e0, e1);
      #pragma unroll
      for (int off = 1; off < 16; off <<= 1)
        tm = fmaxf(tm, __shfl_xor(tm, off, 32));
      const float mnew = fmaxf(mrow[r], tm);
      const float corr = __expf(mrow[r] - mnew);
      const float p0 = __expf(e0 - mnew);
      const float p1 = __expf(e1 - mnew);
      float ps = p0 + p1;
      #pragma unroll
      for (int off = 1; off < 16; off <<= 1)
        ps += __shfl_xor(ps, off, 32);
      lrow[r] = lrow[r] * corr + ps;
      mrow[r] = mnew;
      #pragma unroll
      for (int d4 = 0; d4 < 4; ++d4) acc[d4][r] *= corr;
      myP[(r + hi * 8) * LDSS + l15]      = (_Float16)p0;
      myP[(r + hi * 8) * LDSS + l15 + 16] = (_Float16)p1;
    }
    __syncthreads();   // order P stores before A-fragment gather

    // P as A-fragment (16x32), V^T B-fragments, 4 PV WMMAs
    const _Float16* prp = myP + l15 * LDSS + hi * 8;
    const v16h ap = cat8(*(const v8h*)prp, *(const v8h*)(prp + 16));
    #pragma unroll
    for (int d4 = 0; d4 < 4; ++d4) {
      const _Float16* rp = vp + (size_t)(d4 * 16 + l15) * NSEQ + kt + hi * 8;
      const v16h bv = cat8(*(const v8h*)rp, *(const v8h*)(rp + 16));
      acc[d4] = wmma16(ap, bv, acc[d4]);
    }
  }

  // normalize + store context as [B,N,H*D] f16
  float rinv[8];
  #pragma unroll
  for (int r = 0; r < 8; ++r) rinv[r] = 1.0f / lrow[r];
  #pragma unroll
  for (int d4 = 0; d4 < 4; ++d4)
    #pragma unroll
    for (int r = 0; r < 8; ++r) {
      const int n = q0 + r + hi * 8;
      ctx[((size_t)b * NSEQ + n) * DIM + h * HDIM + d4 * 16 + l15] =
          (_Float16)(acc[d4][r] * rinv[r]);
    }
}

// ===========================================================================
// Kernel 3: output projection + bias.  grid=(1024/128, 4096/128), block=256
// ===========================================================================
__global__ __launch_bounds__(256) void k_out_gemm(
    const _Float16* __restrict__ ctxp, const float* __restrict__ w,
    const float* __restrict__ bias, float* __restrict__ out) {
  __shared__ _Float16 As[128 * LDSS];
  __shared__ _Float16 Bs[128 * LDSS];

  const int tid  = threadIdx.x;
  const int wave = tid >> 5, lane = tid & 31;
  const int l15  = lane & 15, hi = lane >> 4;
  const int waveM = wave >> 2, waveN = wave & 3;
  const int Mtile = blockIdx.y * 128;
  const int Etile = blockIdx.x * 128;

  v8f acc[4][2];
  #pragma unroll
  for (int i = 0; i < 4; ++i)
    #pragma unroll
    for (int j = 0; j < 2; ++j) acc[i][j] = {};

  for (int k0 = 0; k0 < DIM; k0 += 32) {
    #pragma unroll
    for (int p = 0; p < 4; ++p) {
      const int row = p * 32 + (tid >> 3);
      const int cg  = (tid & 7) * 4;
      *(v4h*)(As + row * LDSS + cg) =
          *(const v4h*)(ctxp + (size_t)(Mtile + row) * DIM + k0 + cg);
    }
    {
      const int kk = tid >> 5;
      const int e0 = (tid & 31) * 4;
      #pragma unroll
      for (int p = 0; p < 4; ++p) {
        const int kr = p * 8 + kk;
        float4 wv = *(const float4*)(w + (size_t)(k0 + kr) * DIM + Etile + e0);
        Bs[(e0 + 0) * LDSS + kr] = (_Float16)wv.x;
        Bs[(e0 + 1) * LDSS + kr] = (_Float16)wv.y;
        Bs[(e0 + 2) * LDSS + kr] = (_Float16)wv.z;
        Bs[(e0 + 3) * LDSS + kr] = (_Float16)wv.w;
      }
    }
    __syncthreads();

    v16h af[4], bf[2];
    #pragma unroll
    for (int ms = 0; ms < 4; ++ms)
      af[ms] = lds_afrag(As, waveM * 64 + ms * 16 + l15, hi);
    #pragma unroll
    for (int ns = 0; ns < 2; ++ns)
      bf[ns] = lds_bfrag(Bs, waveN * 32 + ns * 16 + l15, hi);
    #pragma unroll
    for (int ms = 0; ms < 4; ++ms)
      #pragma unroll
      for (int ns = 0; ns < 2; ++ns)
        acc[ms][ns] = wmma16(af[ms], bf[ns], acc[ms][ns]);
    __syncthreads();
  }

  #pragma unroll
  for (int ns = 0; ns < 2; ++ns) {
    const int e = Etile + waveN * 32 + ns * 16 + l15;
    const float be = bias[e];
    #pragma unroll
    for (int ms = 0; ms < 4; ++ms) {
      const int mb = Mtile + waveM * 64 + ms * 16 + hi * 8;
      #pragma unroll
      for (int r = 0; r < 8; ++r)
        out[(size_t)(mb + r) * DIM + e] = acc[ms][ns][r] + be;
    }
  }
}

// ===========================================================================
extern "C" void kernel_launch(void* const* d_in, const int* in_sizes, int n_in,
                              void* d_out, int out_size, void* d_ws, size_t ws_size,
                              hipStream_t stream) {
  const float*         x     = (const float*)d_in[0];
  const unsigned char* mask  = (const unsigned char*)d_in[1];
  const float*         w_qkv = (const float*)d_in[2];
  const float*         w_out = (const float*)d_in[3];
  const float*         b_out = (const float*)d_in[4];
  float*               out   = (float*)d_out;

  const size_t TSZ = (size_t)NB * NHEAD * NSEQ * HDIM;   // 4,194,304 halves
  _Float16* qb  = (_Float16*)d_ws;
  _Float16* kb  = qb + TSZ;
  _Float16* vtb = kb + TSZ;            // stored transposed [B,H,D,N]
  _Float16* ctx = vtb + TSZ;           // [B,N,H*D]
  // total workspace: 4 * TSZ * 2 bytes = 32 MB

  k_qkv_gemm<<<dim3(QKV_E / 128, (NB * NSEQ) / 128), 256, 0, stream>>>(
      x, w_qkv, qb, kb, vtb);
  k_attn<<<dim3(NSEQ / 64, NB * NHEAD), 128, 0, stream>>>(
      qb, kb, vtb, mask, ctx);
  k_out_gemm<<<dim3(DIM / 128, (NB * NSEQ) / 128), 256, 0, stream>>>(
      ctx, w_out, b_out, out);
}
